// DGNN_GA_24146306138480
// MI455X (gfx1250) — matrix operand order repaired and checked by reading, not compiled, not run on hardware
//
#include <hip/hip_runtime.h>
#include <hip/hip_bf16.h>

#define NA 50000
#define NT 50000
#define NE 1000000
#define HD 64

typedef _Float16 half_t;
typedef _Float16 v16h __attribute__((ext_vector_type(16)));
typedef _Float16 v8h  __attribute__((ext_vector_type(8)));
typedef float    v8f  __attribute__((ext_vector_type(8)));

// ---------------------------------------------------------------------------
// WMMA helpers (CDNA5 16x16x32 f16 -> f32)
// ---------------------------------------------------------------------------
static __device__ __forceinline__ v8f wmma_f16(v16h a, v16h b, v8f c) {
  // (neg_a, A, neg_b, B, c_mod, C, reuse_a, reuse_b)
  return __builtin_amdgcn_wmma_f32_16x16x32_f16(false, a, false, b, (short)0, c,
                                                false, false);
}

// A-matrix 16x32 f16 fragment (CDNA5 ISA §7.12.2):
// lane l: m=l%16, h=l/16; VGPR v holds K pair (g*16 + h*8 + (v%4)*2) with g=v/4
// => vector elems 0..7  = row[kbase + h*8 + 0..7]
//    vector elems 8..15 = row[kbase + 16 + h*8 + 0..7]
static __device__ __forceinline__ v16h load_a_frag(const half_t* __restrict__ row,
                                                   int kbase, int h) {
  v8h lo = *(const v8h*)(row + kbase + h * 8);
  v8h hi = *(const v8h*)(row + kbase + 16 + h * 8);
  v16h r;
#pragma unroll
  for (int i = 0; i < 8; ++i) { r[i] = lo[i]; r[i + 8] = hi[i]; }
  return r;
}

// B-matrix 32x16 f16 fragment, B[k][n] = W[n][k] with W row-major [64 x 128]:
// lane l: n=l%16, h=l/16; same K order as A fragment
// => vector elems 0..7 = W[nbase+n][kbase + h*8 + 0..7], 8..15 = W[nbase+n][kbase + 16 + h*8 + 0..7]
static __device__ __forceinline__ v16h load_b_frag(const half_t* sW, int nbase,
                                                   int kbase, int lane) {
  int n = lane & 15, h = lane >> 4;
  const half_t* p = sW + (size_t)(nbase + n) * 128 + kbase + h * 8;
  v8h lo = *(const v8h*)p;
  v8h hi = *(const v8h*)(p + 16);
  v16h r;
#pragma unroll
  for (int i = 0; i < 8; ++i) { r[i] = lo[i]; r[i + 8] = hi[i]; }
  return r;
}

// ---------------------------------------------------------------------------
// Utility kernels
// ---------------------------------------------------------------------------
__global__ void zero_kernel(float* __restrict__ p, int n) {
  int i = blockIdx.x * blockDim.x + threadIdx.x;
  if (i < n) p[i] = 0.0f;
}

// Pack W1 -> f16, and [Wl | Wr] -> f16 K=128 concatenated weights per relation.
__global__ void pack_weights_kernel(const float* __restrict__ Wl_at,
                                    const float* __restrict__ Wr_at,
                                    const float* __restrict__ Wl_ta,
                                    const float* __restrict__ Wr_ta,
                                    const float* __restrict__ W1,
                                    half_t* __restrict__ Wcat_at,
                                    half_t* __restrict__ Wcat_ta,
                                    half_t* __restrict__ W1h) {
  int idx = blockIdx.x * blockDim.x + threadIdx.x;
  if (idx >= 3 * 64 * 128) return;
  int which = idx / 8192;
  int r = idx - which * 8192;
  int j = r >> 7, k = r & 127;
  if (which == 0) {
    W1h[r] = (half_t)W1[r];
  } else {
    const float* Wl = (which == 1) ? Wl_at : Wl_ta;
    const float* Wr = (which == 1) ? Wr_at : Wr_ta;
    half_t* o = (which == 1) ? Wcat_at : Wcat_ta;
    o[r] = (half_t)((k < 64) ? Wl[j * 64 + k] : Wr[j * 64 + (k - 64)]);
  }
}

// Linear -> LayerNorm -> ReLU, one wave per row (lane covers features j, j+32).
__global__ void encoder_kernel(const float* __restrict__ x,
                               const float* __restrict__ W,
                               const float* __restrict__ b,
                               const float* __restrict__ g,
                               const float* __restrict__ beta,
                               half_t* __restrict__ outh, int N, int D) {
  int wave = threadIdx.x >> 5;
  int lane = threadIdx.x & 31;
  int row = blockIdx.x * 4 + wave;
  if (row >= N) return;
  const float* xr = x + (size_t)row * D;
  int j0 = lane, j1 = lane + 32;
  float h0 = b[j0], h1 = b[j1];
  for (int k = 0; k < D; ++k) {
    float xv = xr[k];
    h0 = fmaf(W[j0 * D + k], xv, h0);
    h1 = fmaf(W[j1 * D + k], xv, h1);
  }
  float s = h0 + h1;
#pragma unroll
  for (int m = 1; m < 32; m <<= 1) s += __shfl_xor(s, m, 32);
  float mu = s * (1.0f / 64.0f);
  float d0 = h0 - mu, d1 = h1 - mu;
  float q = d0 * d0 + d1 * d1;
#pragma unroll
  for (int m = 1; m < 32; m <<= 1) q += __shfl_xor(q, m, 32);
  float rs = rsqrtf(q * (1.0f / 64.0f) + 1e-5f);
  float o0 = fmaxf(fmaf(d0 * rs, g[j0], beta[j0]), 0.0f);
  float o1 = fmaxf(fmaf(d1 * rs, g[j1], beta[j1]), 0.0f);
  outh[(size_t)row * HD + j0] = (half_t)o0;
  outh[(size_t)row * HD + j1] = (half_t)o1;
}

// Scatter-sum both directions: 64 threads per edge (one per feature).
__global__ void aggregate_kernel(const half_t* __restrict__ ha,
                                 const half_t* __restrict__ ht,
                                 const long long* __restrict__ ei,
                                 float* __restrict__ sumA, float* __restrict__ sumT,
                                 float* __restrict__ degA, float* __restrict__ degT) {
  int e = blockIdx.x * 4 + (threadIdx.x >> 6);
  int f = threadIdx.x & 63;
  int a = (int)ei[e];
  int t = (int)ei[NE + e];
  float va = (float)ha[(size_t)a * HD + f];
  float vt = (float)ht[(size_t)t * HD + f];
  atomicAdd(&sumT[(size_t)t * HD + f], va);
  atomicAdd(&sumA[(size_t)a * HD + f], vt);
  if (f == 0) {
    atomicAdd(&degT[t], 1.0f);
    atomicAdd(&degA[a], 1.0f);
  }
}

__global__ void normalize_kernel(const float* __restrict__ sum,
                                 const float* __restrict__ deg,
                                 half_t* __restrict__ outh, int N) {
  int i = blockIdx.x * blockDim.x + threadIdx.x;
  if (i >= N * HD) return;
  float c = fmaxf(deg[i >> 6], 1.0f);
  outh[i] = (half_t)(sum[i] / c);
}

// ---------------------------------------------------------------------------
// SAGEConv: out[n] = [m(n) | h(n)] (K=128) @ Wcat^T + bias, stored f16.
// One wave per 16-row tile; Wcat staged in LDS; 16 wmma per tile.
// ---------------------------------------------------------------------------
__global__ void sage_kernel(const half_t* __restrict__ m_h,
                            const half_t* __restrict__ h_h,
                            const half_t* __restrict__ Wcat,
                            const float* __restrict__ bias,
                            half_t* __restrict__ outh, int ntiles) {
  __shared__ __align__(16) half_t sW[64 * 128];
  {
    v8h* d = (v8h*)sW;
    const v8h* s = (const v8h*)Wcat;
    for (int i = threadIdx.x; i < 1024; i += blockDim.x) d[i] = s[i];
  }
  __syncthreads();
  int lane = threadIdx.x & 31;
  int tile = (blockIdx.x * blockDim.x + threadIdx.x) >> 5;
  if (tile >= ntiles) return;
  int row0 = tile * 16;
  int mrow = lane & 15, h = lane >> 4;
  const half_t* r1 = m_h + (size_t)(row0 + mrow) * HD;
  const half_t* r2 = h_h + (size_t)(row0 + mrow) * HD;
  v16h a0 = load_a_frag(r1, 0, h);
  v16h a1 = load_a_frag(r1, 32, h);
  v16h a2 = load_a_frag(r2, 0, h);
  v16h a3 = load_a_frag(r2, 32, h);
#pragma unroll
  for (int nt = 0; nt < 4; ++nt) {
    v8f acc = {};
    acc = wmma_f16(a0, load_b_frag(sW, nt * 16, 0, lane), acc);
    acc = wmma_f16(a1, load_b_frag(sW, nt * 16, 32, lane), acc);
    acc = wmma_f16(a2, load_b_frag(sW, nt * 16, 64, lane), acc);
    acc = wmma_f16(a3, load_b_frag(sW, nt * 16, 96, lane), acc);
    float bcol = bias[nt * 16 + mrow];  // C layout: lane col N = lane%16
#pragma unroll
    for (int r = 0; r < 8; ++r)
      outh[(size_t)(row0 + r + 8 * h) * HD + nt * 16 + mrow] =
          (half_t)(acc[r] + bcol);
  }
}

// ---------------------------------------------------------------------------
// Edge head: z = [ha2[a] | ht2[t]] (K=128); h = relu(z@W1^T + b1);
// logit = h . w2 + b2. One wave per 16 edges; 16 wmma + fused epilogue.
// ---------------------------------------------------------------------------
__global__ void edge_head_kernel(const half_t* __restrict__ ha2,
                                 const half_t* __restrict__ ht2,
                                 const long long* __restrict__ ei,
                                 const half_t* __restrict__ W1h,
                                 const float* __restrict__ b1,
                                 const float* __restrict__ W2,
                                 const float* __restrict__ b2,
                                 float* __restrict__ out, int ntiles) {
  __shared__ __align__(16) half_t sW[64 * 128];
  {
    v8h* d = (v8h*)sW;
    const v8h* s = (const v8h*)W1h;
    for (int i = threadIdx.x; i < 1024; i += blockDim.x) d[i] = s[i];
  }
  __syncthreads();
  int lane = threadIdx.x & 31;
  int tile = (blockIdx.x * blockDim.x + threadIdx.x) >> 5;
  if (tile >= ntiles) return;
  int e0 = tile * 16;
  int n = lane & 15, h = lane >> 4;  // n doubles as A-row (edge) and C-col
  int e = e0 + n;
  int a = (int)ei[e];
  int t = (int)ei[NE + e];
  const half_t* ra = ha2 + (size_t)a * HD;
  const half_t* rt = ht2 + (size_t)t * HD;
  v16h a0 = load_a_frag(ra, 0, h);
  v16h a1 = load_a_frag(ra, 32, h);
  v16h a2 = load_a_frag(rt, 0, h);
  v16h a3 = load_a_frag(rt, 32, h);
  float wsum[8];
#pragma unroll
  for (int r = 0; r < 8; ++r) wsum[r] = 0.0f;
#pragma unroll
  for (int nt = 0; nt < 4; ++nt) {
    v8f acc = {};
    acc = wmma_f16(a0, load_b_frag(sW, nt * 16, 0, lane), acc);
    acc = wmma_f16(a1, load_b_frag(sW, nt * 16, 32, lane), acc);
    acc = wmma_f16(a2, load_b_frag(sW, nt * 16, 64, lane), acc);
    acc = wmma_f16(a3, load_b_frag(sW, nt * 16, 96, lane), acc);
    float b1c = b1[nt * 16 + n];
    float w2c = W2[nt * 16 + n];
#pragma unroll
    for (int r = 0; r < 8; ++r) {
      float v = fmaxf(acc[r] + b1c, 0.0f);
      wsum[r] = fmaf(v, w2c, wsum[r]);
    }
  }
  // reduce over the 64 hidden features: sum across lanes within each 16-group
#pragma unroll
  for (int mm = 1; mm < 16; mm <<= 1) {
#pragma unroll
    for (int r = 0; r < 8; ++r) wsum[r] += __shfl_xor(wsum[r], mm, 32);
  }
  if (n == 0) {
    float bb = b2[0];
#pragma unroll
    for (int r = 0; r < 8; ++r) out[e0 + r + 8 * h] = wsum[r] + bb;
  }
}

// ---------------------------------------------------------------------------
extern "C" void kernel_launch(void* const* d_in, const int* in_sizes, int n_in,
                              void* d_out, int out_size, void* d_ws, size_t ws_size,
                              hipStream_t stream) {
  (void)in_sizes; (void)n_in; (void)out_size; (void)ws_size;
  const float*     x_agent = (const float*)d_in[0];
  const float*     x_task  = (const float*)d_in[1];
  const long long* ei      = (const long long*)d_in[2];
  const float* Wa = (const float*)d_in[3];
  const float* ba = (const float*)d_in[4];
  const float* ga = (const float*)d_in[5];
  const float* bga = (const float*)d_in[6];
  const float* Wt = (const float*)d_in[7];
  const float* bt = (const float*)d_in[8];
  const float* gt = (const float*)d_in[9];
  const float* bgt = (const float*)d_in[10];
  const float* Wl_at = (const float*)d_in[11];
  const float* bl_at = (const float*)d_in[12];
  const float* Wr_at = (const float*)d_in[13];
  const float* Wl_ta = (const float*)d_in[14];
  const float* bl_ta = (const float*)d_in[15];
  const float* Wr_ta = (const float*)d_in[16];
  const float* W1 = (const float*)d_in[17];
  const float* b1 = (const float*)d_in[18];
  const float* W2 = (const float*)d_in[19];
  const float* b2 = (const float*)d_in[20];
  float* logits = (float*)d_out;

  char* ws = (char*)d_ws;
  size_t off = 0;
  auto carve = [&](size_t bytes) -> void* {
    void* p = (void*)(ws + off);
    off += (bytes + 255) & ~(size_t)255;
    return p;
  };
  half_t* ha_h   = (half_t*)carve((size_t)NA * HD * sizeof(half_t));
  half_t* ht_h   = (half_t*)carve((size_t)NT * HD * sizeof(half_t));
  half_t* m_a_h  = (half_t*)carve((size_t)NA * HD * sizeof(half_t));
  half_t* m_t_h  = (half_t*)carve((size_t)NT * HD * sizeof(half_t));
  half_t* ha2_h  = (half_t*)carve((size_t)NA * HD * sizeof(half_t));
  half_t* ht2_h  = (half_t*)carve((size_t)NT * HD * sizeof(half_t));
  half_t* W1h     = (half_t*)carve((size_t)64 * 128 * sizeof(half_t));
  half_t* Wcat_at = (half_t*)carve((size_t)64 * 128 * sizeof(half_t));
  half_t* Wcat_ta = (half_t*)carve((size_t)64 * 128 * sizeof(half_t));
  size_t zero_begin = off;
  float* sumA = (float*)carve((size_t)NA * HD * sizeof(float));
  float* sumT = (float*)carve((size_t)NT * HD * sizeof(float));
  float* degA = (float*)carve((size_t)NA * sizeof(float));
  float* degT = (float*)carve((size_t)NT * sizeof(float));
  size_t zero_end = off;
  int nzero = (int)((zero_end - zero_begin) / sizeof(float));

  // 1) clear accumulators (includes alignment padding; deterministic init)
  zero_kernel<<<(nzero + 255) / 256, 256, 0, stream>>>((float*)(ws + zero_begin), nzero);

  // 2) pack f16 weights
  pack_weights_kernel<<<(3 * 64 * 128 + 255) / 256, 256, 0, stream>>>(
      Wl_at, Wr_at, Wl_ta, Wr_ta, W1, Wcat_at, Wcat_ta, W1h);

  // 3) node encoders (Linear + LayerNorm + ReLU -> f16)
  encoder_kernel<<<NA / 4, 128, 0, stream>>>(x_agent, Wa, ba, ga, bga, ha_h, NA, 32);
  encoder_kernel<<<NT / 4, 128, 0, stream>>>(x_task, Wt, bt, gt, bgt, ht_h, NT, 48);

  // 4) scatter-sum both directions + degrees
  aggregate_kernel<<<NE / 4, 256, 0, stream>>>(ha_h, ht_h, ei, sumA, sumT, degA, degT);

  // 5) mean + f16 cast
  normalize_kernel<<<(NT * HD + 255) / 256, 256, 0, stream>>>(sumT, degT, m_t_h, NT);
  normalize_kernel<<<(NA * HD + 255) / 256, 256, 0, stream>>>(sumA, degA, m_a_h, NA);

  // 6) SAGE convolutions (WMMA, K=128 folded)
  const int sage_tiles = NT / 16;  // 3125
  const int sage_blocks = (sage_tiles * 32 + 127) / 128;
  sage_kernel<<<sage_blocks, 128, 0, stream>>>(m_t_h, ht_h, Wcat_at, bl_at, ht2_h, sage_tiles);
  sage_kernel<<<sage_blocks, 128, 0, stream>>>(m_a_h, ha_h, Wcat_ta, bl_ta, ha2_h, sage_tiles);

  // 7) edge head (gathered WMMA GEMM + fused MLP epilogue)
  const int edge_tiles = NE / 16;  // 62500
  edge_head_kernel<<<(edge_tiles * 32) / 128, 128, 0, stream>>>(
      ha2_h, ht2_h, ei, W1h, b1, W2, b2, logits, edge_tiles);
}
